// MoEFFN_68728066671341
// MI455X (gfx1250) — matrix operand order repaired and checked by reading, not compiled, not run on hardware
//
#include <hip/hip_runtime.h>
#include <hip/hip_bf16.h>

// ---------------------------------------------------------------------------
// MoE top-1 FFN for MI455X (gfx1250, wave32, WMMA).
//   S=2048 N=8 D=1024 F=4096 E=8 ; T = S*N = 16384 tokens
// Pipeline:
//   k_init      : zero per-expert counters
//   k_gate      : gate scores + softmax top1 (one wave per token),
//                 also converts x -> bf16 (xb)
//   k_transcvt  : fp32 -> bf16 convert + transpose of W1 (-> [E][D][F]) and
//                 W2 (-> [E][F][D]) so WMMA B-fragments are contiguous
//   k_scan      : exclusive prefix over expert counts
//   k_scatter   : build per-expert token lists
//   k_ffn       : 32-token tile per block, 8 waves; each wave computes 32x64
//                 register tiles (acc[2][4], 8 WMMAs per k-step) to keep
//                 register pressure bounded (no scratch spills).
// ---------------------------------------------------------------------------

typedef __attribute__((ext_vector_type(16))) __bf16 v16bf;
typedef __attribute__((ext_vector_type(8)))  __bf16 v8bf;
typedef __attribute__((ext_vector_type(8)))  float  v8f;

#define SEQ   2048
#define NH    8
#define DDIM  1024
#define FDIM  4096
#define NEXP  8
#define NTOK  (SEQ * NH)          // 16384
#define TM    32                  // tokens per ffn tile
#define MAXT  (NTOK / TM)         // 512 tiles max per expert
#define SH    (FDIM + 8)          // LDS H-tile row stride in halves (bank-friendly)
#define HBYTES (TM * SH * 2)      // 262,656 bytes
#define FFN_SMEM (HBYTES + 256)   // + token ids + probs

static __device__ inline v8f wmma_bf16(v16bf a, v16bf b, v8f c) {
  // emits v_wmma_f32_16x16x32_bf16
  return __builtin_amdgcn_wmma_f32_16x16x32_bf16(
      /*neg_a=*/false, a, /*neg_b=*/false, b,
      /*c_mod=*/(short)0, c, /*reuse_a=*/false, /*reuse_b=*/false);
}

static __device__ inline v16bf cat16(v8bf lo, v8bf hi) {
  return __builtin_shufflevector(lo, hi, 0,1,2,3,4,5,6,7,8,9,10,11,12,13,14,15);
}

static __device__ inline float gelu_exact(float x) {
  return 0.5f * x * (1.0f + erff(x * 0.70710678118654752440f));
}

// ------------------------------- init --------------------------------------
__global__ void k_init(int* __restrict__ cnt) {
  if (threadIdx.x < NEXP) cnt[threadIdx.x] = 0;
}

// ------------------------------- gating ------------------------------------
// One wave per token: 8 expert dot products over D, softmax, argmax(first),
// and bf16 conversion of the token row.
__global__ __launch_bounds__(128) void
k_gate(const float* __restrict__ x, const float* __restrict__ Wg,
       __bf16* __restrict__ xb, int* __restrict__ eidx,
       float* __restrict__ prob, int* __restrict__ cnt) {
  const int t    = blockIdx.x * 4 + (threadIdx.x >> 5);
  const int lane = threadIdx.x & 31;
  const float* xr = x + (size_t)t * DDIM;
  __bf16* xo = xb + (size_t)t * DDIM;

  float acc[NEXP];
#pragma unroll
  for (int e = 0; e < NEXP; ++e) acc[e] = 0.0f;

  for (int d = lane; d < DDIM; d += 32) {
    float xv = xr[d];
    xo[d] = (__bf16)xv;
#pragma unroll
    for (int e = 0; e < NEXP; ++e) acc[e] += xv * Wg[e * DDIM + d];
  }
#pragma unroll
  for (int e = 0; e < NEXP; ++e)
#pragma unroll
    for (int o = 16; o > 0; o >>= 1) acc[e] += __shfl_xor(acc[e], o, 32);

  if (lane == 0) {
    int best = 0; float bs = acc[0];
#pragma unroll
    for (int e = 1; e < NEXP; ++e)
      if (acc[e] > bs) { bs = acc[e]; best = e; }   // strict > : first max wins
    float s = 0.0f;
#pragma unroll
    for (int e = 0; e < NEXP; ++e) s += expf(acc[e] - bs);
    eidx[t] = best;
    prob[t] = 1.0f / s;                              // softmax prob of argmax
    atomicAdd(&cnt[best], 1);
  }
}

// ------------------- weight convert + transpose (fp32 -> bf16) -------------
// in : per-expert [R][C] fp32 ; out : per-expert [C][R] bf16
__global__ __launch_bounds__(256) void
k_transcvt(const float* __restrict__ in, __bf16* __restrict__ outp,
           int R, int C) {
  __shared__ float tile[32][33];
  const int ez = blockIdx.z;
  const size_t base = (size_t)ez * (size_t)R * (size_t)C;
  const int cb = blockIdx.x * 32, rb = blockIdx.y * 32;
  const int tx = threadIdx.x, ty = threadIdx.y;
#pragma unroll
  for (int i = 0; i < 4; ++i) {
    int r = rb + ty + i * 8;
    tile[ty + i * 8][tx] = in[base + (size_t)r * C + cb + tx];
  }
  __syncthreads();
#pragma unroll
  for (int i = 0; i < 4; ++i) {
    int c = cb + ty + i * 8;
    outp[base + (size_t)c * R + rb + tx] = (__bf16)tile[tx][ty + i * 8];
  }
}

// ------------------------------- prefix scan -------------------------------
__global__ void k_scan(const int* __restrict__ cnt, int* __restrict__ offs,
                       int* __restrict__ cursor) {
  if (threadIdx.x == 0) {
    int run = 0;
#pragma unroll
    for (int e = 0; e < NEXP; ++e) { offs[e] = run; run += cnt[e]; }
  }
  if (threadIdx.x < NEXP) cursor[threadIdx.x] = 0;
}

// ------------------------------- scatter -----------------------------------
__global__ __launch_bounds__(256) void
k_scatter(const int* __restrict__ eidx, const int* __restrict__ offs,
          int* __restrict__ cursor, int* __restrict__ order) {
  const int t = blockIdx.x * 256 + threadIdx.x;
  const int e = eidx[t];
  const int pos = offs[e] + atomicAdd(&cursor[e], 1);
  order[pos] = t;
}

// ------------------------------- expert FFN --------------------------------
// Grid: E * MAXT blocks; block = 256 threads (8 wave32).
// Per wave: accumulator-stationary 32x64 register tile; 8 WMMAs per k-step.
__global__ __launch_bounds__(256) void
k_ffn(const __bf16* __restrict__ xb,
      const __bf16* __restrict__ W1t,   // [E][D][F] bf16
      const __bf16* __restrict__ W2t,   // [E][F][D] bf16
      const float*  __restrict__ b1,    // [E][F]
      const float*  __restrict__ b2,    // [E][D]
      const int*    __restrict__ order,
      const int*    __restrict__ offs,
      const int*    __restrict__ cnt,
      const float*  __restrict__ prob,
      float*        __restrict__ out) {
  extern __shared__ __align__(16) char smem[];
  __bf16* lds_h    = (__bf16*)smem;                       // [TM][SH] bf16
  int*    lds_tok  = (int*)  (smem + HBYTES);             // [TM]
  float*  lds_prob = (float*)(smem + HBYTES + 128);       // [TM]

  const int e    = blockIdx.x >> 9;       // MAXT == 512
  const int tile = blockIdx.x & (MAXT - 1);
  const int ne   = cnt[e];
  const int tb   = tile * TM;
  if (tb >= ne) return;                   // uniform per block
  const int nrows = min(TM, ne - tb);
  const int base  = offs[e] + tb;

  const int tid = threadIdx.x;
  if (tid < TM) {
    int idx = (tid < nrows) ? tid : 0;    // duplicate row 0 for padding
    int t = order[base + idx];
    lds_tok[tid]  = t;
    lds_prob[tid] = prob[t];
  }
  __syncthreads();

  const int lane = tid & 31;
  const int wave = tid >> 5;
  const int half = lane >> 4;             // hi/lo half-wave
  const int ln   = lane & 15;
  const int koff = half * 8;              // A-frag K sub-offset (halves)

  // A rows: lane ln holds row ln (tile0) and row ln+16 (tile1)
  const __bf16* pA0 = xb + (size_t)lds_tok[ln]      * DDIM;
  const __bf16* pA1 = xb + (size_t)lds_tok[ln + 16] * DDIM;
  // B for GEMM1: lane holds column f = ln of W1t[e], K rows in table order
  const __bf16* pB1 = W1t + (size_t)e * DDIM * FDIM + ln;

  // ---------------- GEMM1: H[32,F] = gelu(X @ W1^T + b1) -------------------
  // Wave owns 512 f-columns = 8 blocks of 64.
#pragma unroll 1
  for (int fblk = 0; fblk < 8; ++fblk) {
    const int fb = wave * 512 + fblk * 64;
    v8f acc[2][4];
#pragma unroll
    for (int mi = 0; mi < 2; ++mi)
#pragma unroll
      for (int j = 0; j < 4; ++j) acc[mi][j] = {};

#pragma unroll 2
    for (int kb = 0; kb < DDIM; kb += 32) {
      v16bf a0 = cat16(*(const v8bf*)(pA0 + kb + koff),
                       *(const v8bf*)(pA0 + kb + koff + 16));
      v16bf a1 = cat16(*(const v8bf*)(pA1 + kb + koff),
                       *(const v8bf*)(pA1 + kb + koff + 16));
      const __bf16* bp = pB1 + (size_t)kb * FDIM + fb;
#pragma unroll
      for (int j = 0; j < 4; ++j) {
        v16bf b;
#pragma unroll
        for (int s = 0; s < 16; ++s)
          b[s] = bp[(size_t)(koff + (s & 7) + (s >> 3) * 16) * FDIM + j * 16];
        acc[0][j] = wmma_bf16(a0, b, acc[0][j]);
        acc[1][j] = wmma_bf16(a1, b, acc[1][j]);
      }
    }
    // epilogue: bias + gelu -> bf16 H tile in LDS
#pragma unroll
    for (int j = 0; j < 4; ++j) {
      const int col = fb + j * 16 + ln;
      const float bias = b1[(size_t)e * FDIM + col];
#pragma unroll
      for (int i = 0; i < 8; ++i) {
        const int m0 = i + half * 8;
        lds_h[m0 * SH + col]        = (__bf16)gelu_exact(acc[0][j][i] + bias);
        lds_h[(m0 + 16) * SH + col] = (__bf16)gelu_exact(acc[1][j][i] + bias);
      }
    }
  }
  __syncthreads();

  // ---------------- GEMM2: Y[32,D] = (H @ W2^T + b2) * prob ----------------
  const __bf16* pB2 = W2t + (size_t)e * FDIM * DDIM + ln;
  const __bf16* pH0 = lds_h + ln * SH;
  const __bf16* pH1 = lds_h + (ln + 16) * SH;
  // Wave owns 128 d-columns = 2 blocks of 64.
#pragma unroll 1
  for (int dblk = 0; dblk < 2; ++dblk) {
    const int db = wave * 128 + dblk * 64;
    v8f acc[2][4];
#pragma unroll
    for (int mi = 0; mi < 2; ++mi)
#pragma unroll
      for (int j = 0; j < 4; ++j) acc[mi][j] = {};

#pragma unroll 2
    for (int kb = 0; kb < FDIM; kb += 32) {
      v16bf a0 = cat16(*(const v8bf*)(pH0 + kb + koff),
                       *(const v8bf*)(pH0 + kb + koff + 16));
      v16bf a1 = cat16(*(const v8bf*)(pH1 + kb + koff),
                       *(const v8bf*)(pH1 + kb + koff + 16));
      const __bf16* bp = pB2 + (size_t)kb * DDIM + db;
#pragma unroll
      for (int j = 0; j < 4; ++j) {
        v16bf b;
#pragma unroll
        for (int s = 0; s < 16; ++s)
          b[s] = bp[(size_t)(koff + (s & 7) + (s >> 3) * 16) * DDIM + j * 16];
        acc[0][j] = wmma_bf16(a0, b, acc[0][j]);
        acc[1][j] = wmma_bf16(a1, b, acc[1][j]);
      }
    }
    // epilogue: bias + prob scale, scatter rows back to y
#pragma unroll
    for (int j = 0; j < 4; ++j) {
      const int col = db + j * 16 + ln;
      const float bias = b2[(size_t)e * DDIM + col];
#pragma unroll
      for (int i = 0; i < 8; ++i) {
        const int m0 = i + half * 8;
        if (m0 < nrows)
          out[(size_t)lds_tok[m0] * DDIM + col] =
              (acc[0][j][i] + bias) * lds_prob[m0];
        const int m1 = m0 + 16;
        if (m1 < nrows)
          out[(size_t)lds_tok[m1] * DDIM + col] =
              (acc[1][j][i] + bias) * lds_prob[m1];
      }
    }
  }
}

// ------------------------------- launcher ----------------------------------
extern "C" void kernel_launch(void* const* d_in, const int* in_sizes, int n_in,
                              void* d_out, int out_size, void* d_ws, size_t ws_size,
                              hipStream_t stream) {
  const float* x  = (const float*)d_in[0];
  const float* Wg = (const float*)d_in[1];
  const float* W1 = (const float*)d_in[2];
  const float* b1 = (const float*)d_in[3];
  const float* W2 = (const float*)d_in[4];
  const float* b2 = (const float*)d_in[5];
  float* out = (float*)d_out;

  // workspace carve-up (all sizes 256B-aligned)
  char* ws = (char*)d_ws;
  __bf16* xb   = (__bf16*)ws;  ws += (size_t)NTOK * DDIM * 2;          // 32 MiB
  __bf16* W1t  = (__bf16*)ws;  ws += (size_t)NEXP * DDIM * FDIM * 2;   // 64 MiB
  __bf16* W2t  = (__bf16*)ws;  ws += (size_t)NEXP * FDIM * DDIM * 2;   // 64 MiB
  float* prob  = (float*)ws;   ws += (size_t)NTOK * 4;
  int*   eidx  = (int*)ws;     ws += (size_t)NTOK * 4;
  int*   order = (int*)ws;     ws += (size_t)NTOK * 4;
  int*   cnt   = (int*)ws;     ws += 256;
  int*   offs  = (int*)ws;     ws += 256;
  int*   cursor= (int*)ws;     ws += 256;

  k_init<<<1, 32, 0, stream>>>(cnt);
  k_gate<<<NTOK / 4, 128, 0, stream>>>(x, Wg, xb, eidx, prob, cnt);
  // W1: [E][F][D] -> W1t [E][D][F]
  k_transcvt<<<dim3(DDIM / 32, FDIM / 32, NEXP), dim3(32, 8), 0, stream>>>(W1, W1t, FDIM, DDIM);
  // W2: [E][D][F] -> W2t [E][F][D]
  k_transcvt<<<dim3(FDIM / 32, DDIM / 32, NEXP), dim3(32, 8), 0, stream>>>(W2, W2t, DDIM, FDIM);
  k_scan<<<1, 32, 0, stream>>>(cnt, offs, cursor);
  k_scatter<<<NTOK / 256, 256, 0, stream>>>(eidx, offs, cursor, order);
  k_ffn<<<NEXP * MAXT, 256, FFN_SMEM, stream>>>(xb, W1t, W2t, b1, b2,
                                                order, offs, cnt, prob, out);
}
